// CausalSelfAttention_82523501625390
// MI455X (gfx1250) — matrix operand order repaired and checked by reading, not compiled, not run on hardware
//
#include <hip/hip_runtime.h>
#include <hip/hip_bf16.h>

// ---------------------------------------------------------------------------
// MI455X (gfx1250) causal self-attention, bf16 WMMA path.
//   B=128, T=256, C=384, H=6, hd=64.
// Pipeline: f32->bf16 casts -> qkv GEMM (wmma bf16) -> per-(b,h) flash
// attention (wmma bf16; f32 online softmax: row-max via v_permlane16
// butterflies, row-sum via an extra WMMA against an all-ones fragment) ->
// proj GEMM (wmma bf16, f32 out).
// ---------------------------------------------------------------------------

typedef __attribute__((ext_vector_type(16))) __bf16        v16bf;
typedef __attribute__((ext_vector_type(8)))  float         v8f;
typedef __attribute__((ext_vector_type(4)))  unsigned int  u32x4;
typedef __attribute__((ext_vector_type(2)))  unsigned int  u32x2;
typedef __attribute__((ext_vector_type(4)))  float         f32x4;

union Frag16 { v16bf v; u32x4 u[2]; };

__device__ __forceinline__ unsigned short f2bf(float f) {
  unsigned int u = __float_as_uint(f);
  u += 0x7FFFu + ((u >> 16) & 1u);          // round-to-nearest-even
  return (unsigned short)(u >> 16);
}

// A-matrix 16x32 bf16 fragment (ISA 7.12.2): lane holds row M=lane&15,
// halfwords 0..7 = K kb..kb+7, halfwords 8..15 = K kb+16..kb+23, kb=(lane>>4)*8.
__device__ __forceinline__ v16bf load_a16(const unsigned short* p, int hf) {
  Frag16 f;
  f.u[0] = *reinterpret_cast<const u32x4*>(p + hf * 8);
  f.u[1] = *reinterpret_cast<const u32x4*>(p + hf * 8 + 16);
  return f.v;
}

// B-matrix 32x16 bf16 fragment: lane holds column N=lane&15,
// halfwords 0..7 = K kb..kb+7, halfwords 8..15 = K kb+16..kb+23, kb=(lane>>4)*8.
__device__ __forceinline__ v16bf load_b16(const unsigned short* p, int hf) {
  Frag16 f;
  f.u[0] = *reinterpret_cast<const u32x4*>(p + hf * 8);
  f.u[1] = *reinterpret_cast<const u32x4*>(p + hf * 8 + 16);
  return f.v;
}

// All-ones bf16 B fragment (layout-independent): used for WMMA row sums.
__device__ __forceinline__ v16bf ones_frag() {
  Frag16 f;
  u32x4 k = {0x3F803F80u, 0x3F803F80u, 0x3F803F80u, 0x3F803F80u};
  f.u[0] = k;
  f.u[1] = k;
  return f.v;
}

__device__ __forceinline__ v8f wmma_bf16(v16bf a, v16bf b, v8f c) {
  return __builtin_amdgcn_wmma_f32_16x16x32_bf16(false, a, false, b,
                                                 (short)0, c, false, false);
}

// Max over each 16-lane group, butterfly.  v_permlane16_b32 (pure VALU,
// no LDS round trip) when available; ds_bpermute fallback otherwise.
#if __has_builtin(__builtin_amdgcn_permlane16)
__device__ __forceinline__ float pl16(float v, unsigned s0, unsigned s1) {
  unsigned u = __float_as_uint(v);
  u = __builtin_amdgcn_permlane16(u, u, s0, s1, false, false);
  return __uint_as_float(u);
}
__device__ __forceinline__ float lane16_max(float v) {
  v = fmaxf(v, pl16(v, 0x67452301u, 0xEFCDAB89u));   // xor 1
  v = fmaxf(v, pl16(v, 0x54761032u, 0xDCFE98BAu));   // xor 2
  v = fmaxf(v, pl16(v, 0x32107654u, 0xBA98FEDCu));   // xor 4
  v = fmaxf(v, pl16(v, 0xFEDCBA98u, 0x76543210u));   // xor 8
  return v;
}
#else
__device__ __forceinline__ float lane16_max(float v) {
  v = fmaxf(v, __shfl_xor(v, 8, 32));
  v = fmaxf(v, __shfl_xor(v, 4, 32));
  v = fmaxf(v, __shfl_xor(v, 2, 32));
  v = fmaxf(v, __shfl_xor(v, 1, 32));
  return v;
}
#endif

// ---------------------------------------------------------------------------
// f32 -> bf16 cast, 4 elements per thread
// ---------------------------------------------------------------------------
__global__ __launch_bounds__(256) void cvt_f32_bf16(
    const float* __restrict__ in, unsigned short* __restrict__ out, int n4) {
  int i = blockIdx.x * 256 + threadIdx.x;
  if (i >= n4) return;
  f32x4 f = reinterpret_cast<const f32x4*>(in)[i];
  u32x2 o;
  o.x = (unsigned)f2bf(f.x) | ((unsigned)f2bf(f.y) << 16);
  o.y = (unsigned)f2bf(f.z) | ((unsigned)f2bf(f.w) << 16);
  reinterpret_cast<u32x2*>(out)[i] = o;
}

// ---------------------------------------------------------------------------
// C[M,N] = A[M,K](bf16) * W[N,K]^T(bf16) + bias.  One wave -> 16x128 tile
// (8 n-tiles, A fragment reused 8x).  Block = 8 waves.
// ---------------------------------------------------------------------------
template <bool F32OUT>
__global__ __launch_bounds__(256) void gemm_bias_wmma(
    const unsigned short* __restrict__ A,
    const unsigned short* __restrict__ W,
    const float* __restrict__ bias,
    unsigned short* __restrict__ outb,
    float* __restrict__ outf,
    int M, int N, int K) {
  const int lane = threadIdx.x & 31;
  const int wave = threadIdx.x >> 5;
  const int hf   = lane >> 4;
  const int ln   = lane & 15;
  const int nG   = N >> 7;                 // 128-wide column groups
  const int t    = blockIdx.x * 8 + wave;
  const int tm   = t / nG;
  const int tn   = t - tm * nG;
  const int colBase = tn * 128;
  const int row  = tm * 16 + ln;

  v8f c[8];
#pragma unroll
  for (int nt = 0; nt < 8; ++nt) {
    float bv = bias[colBase + nt * 16 + ln];
#pragma unroll
    for (int i = 0; i < 8; ++i) c[nt][i] = bv;
  }

  const unsigned short* arow = A + (size_t)row * K;
  for (int kk = 0; kk < K; kk += 32) {
    if (kk + 32 < K)                       // gfx1250 global_prefetch_b8
      __builtin_prefetch(arow + kk + 32, 0, 1);
    v16bf a = load_a16(arow + kk, hf);
#pragma unroll
    for (int nt = 0; nt < 8; ++nt) {
      const unsigned short* wrow = W + (size_t)(colBase + nt * 16 + ln) * K + kk;
      v16bf b = load_b16(wrow, hf);
      c[nt] = wmma_bf16(a, b, c[nt]);
    }
  }

#pragma unroll
  for (int nt = 0; nt < 8; ++nt) {
#pragma unroll
    for (int r = 0; r < 8; ++r) {
      int orow = tm * 16 + r + hf * 8;          // C-layout: lanes>=16 -> M+8
      int ocol = colBase + nt * 16 + ln;
      size_t idx = (size_t)orow * N + ocol;
      if (F32OUT) outf[idx] = c[nt][r];
      else        outb[idx] = f2bf(c[nt][r]);
    }
  }
}

// ---------------------------------------------------------------------------
// One 32-key block of flash attention for one wave (2 query m-tiles).
// MASKED is compile-time: only the diagonal block pays the causal cndmasks.
// ---------------------------------------------------------------------------
template <bool MASKED>
__device__ __forceinline__ void attn_kblock(
    const unsigned short* __restrict__ base,   // qkv row base for this batch
    unsigned short* VT,                        // LDS V^T, stride 272
    unsigned short* pwave,                     // per-wave P staging (2 x 512)
    const v16bf* qa,                           // [4] = qa[mt*2 + kstep]
    v16bf onesf,                               // all-ones B fragment
    float* rm, float* rs,                      // [16] = [mt*8 + r]
    v8f* yacc,                                 // [8]  = yacc[mt*4 + d]
    int h, int jp, int m0, int hf, int ln) {
  // K B-fragments: column n = key, contiguous over d.
  const unsigned short* kr0 = base + (size_t)(jp * 32 + ln) * 1152 + 384 + h * 64;
  const unsigned short* kr1 = kr0 + (size_t)16 * 1152;
  v16bf kb00 = load_b16(kr0, hf);        // keys j0, d 0..31
  v16bf kb01 = load_b16(kr0 + 32, hf);   // keys j0, d 32..63
  v16bf kb10 = load_b16(kr1, hf);
  v16bf kb11 = load_b16(kr1 + 32, hf);
  // V B-fragments from LDS: column n = d0+n, contiguous over key.
  v16bf vb[4];
#pragma unroll
  for (int d = 0; d < 4; ++d)
    vb[d] = load_b16(&VT[(d * 16 + ln) * 272 + jp * 32], hf);

#pragma unroll
  for (int mt = 0; mt < 2; ++mt) {
    v8f s0 = {};
    v8f s1 = {};
    s0 = wmma_bf16(qa[mt * 2 + 0], kb00, s0);
    s0 = wmma_bf16(qa[mt * 2 + 1], kb01, s0);
    s1 = wmma_bf16(qa[mt * 2 + 0], kb10, s1);
    s1 = wmma_bf16(qa[mt * 2 + 1], kb11, s1);

    unsigned short* pbuf = pwave + mt * 512;
    float corr[8];
#pragma unroll
    for (int r = 0; r < 8; ++r) {
      float v0 = s0[r] * 0.125f;           // 1/sqrt(64)
      float v1 = s1[r] * 0.125f;
      if (MASKED) {
        int rowg = m0 + mt * 16 + r + hf * 8;
        int c0 = jp * 32 + ln;
        if (c0 > rowg)      v0 = -1e30f;
        if (c0 + 16 > rowg) v1 = -1e30f;
      }
      int si = mt * 8 + r;
      float bmax = lane16_max(fmaxf(v0, v1));
      float nm   = fmaxf(rm[si], bmax);
      corr[r]    = __expf(rm[si] - nm);
      rm[si] = nm;
      float p0 = __expf(v0 - nm);
      float p1 = __expf(v1 - nm);
#pragma unroll
      for (int d = 0; d < 4; ++d) yacc[mt * 4 + d][r] *= corr[r];
      int prow = r + hf * 8;
      pbuf[prow * 32 + ln]      = f2bf(p0);
      pbuf[prow * 32 + 16 + ln] = f2bf(p1);
    }
    // Re-layout P (C-frag) -> A-frag through per-wave LDS tile.
    v16bf pa = load_a16(pbuf + ln * 32, hf);
    // Row sums of this P block via WMMA against all-ones B (sums exactly
    // the bf16 P that feeds P@V).
    v8f ssum = {};
    ssum = wmma_bf16(pa, onesf, ssum);
#pragma unroll
    for (int r = 0; r < 8; ++r)
      rs[mt * 8 + r] = rs[mt * 8 + r] * corr[r] + ssum[r];
    // P@V accumulation.
#pragma unroll
    for (int d = 0; d < 4; ++d)
      yacc[mt * 4 + d] = wmma_bf16(pa, vb[d], yacc[mt * 4 + d]);
  }
}

// ---------------------------------------------------------------------------
// Flash attention, one workgroup per (b, head). 8 waves x 32 query rows.
// qkv: [B, T, 1152] bf16 (q|k|v each 384 cols). y: [B, T, 384] bf16.
// ---------------------------------------------------------------------------
__global__ __launch_bounds__(256) void attn_kernel(
    const unsigned short* __restrict__ qkv, unsigned short* __restrict__ y) {
  __shared__ __align__(32) unsigned short VT[64 * 272];        // V^T, padded
  __shared__ __align__(16) unsigned short Pb[8][2][16 * 32];   // per-wave P

  const int bh = blockIdx.x;
  const int bb = bh / 6;
  const int h  = bh % 6;
  const int tid  = threadIdx.x;
  const int wave = tid >> 5;
  const int lane = tid & 31;
  const int hf   = lane >> 4;
  const int ln   = lane & 15;

  const unsigned short* base = qkv + (size_t)bb * 256 * 1152;

  // Stage V transposed into LDS: VT[d][key], row stride 272 (32B-aligned frags)
  for (int i = tid; i < 256 * 64; i += 256) {
    int key = i >> 6, d = i & 63;
    VT[d * 272 + key] = base[(size_t)key * 1152 + 768 + h * 64 + d];
  }
  __syncthreads();

  const int m0 = wave * 32;

  // Hoist Q A-fragments (2 m-tiles x 2 K-steps over d=64)
  v16bf qa[4];
#pragma unroll
  for (int mt = 0; mt < 2; ++mt) {
    int row = m0 + mt * 16 + ln;
    const unsigned short* qr = base + (size_t)row * 1152 + h * 64;
    qa[mt * 2 + 0] = load_a16(qr, hf);
    qa[mt * 2 + 1] = load_a16(qr + 32, hf);
  }
  v16bf onesf = ones_frag();

  float rm[16], rs[16];
  v8f yacc[8];
#pragma unroll
  for (int i = 0; i < 16; ++i) { rm[i] = -1e30f; rs[i] = 0.f; }
#pragma unroll
  for (int f = 0; f < 8; ++f)
#pragma unroll
    for (int i = 0; i < 8; ++i) yacc[f][i] = 0.f;

  unsigned short* pwave = &Pb[wave][0][0];

  // Causal: wave w's rows are [32w, 32w+32).  Blocks 0..w-1 are fully
  // unmasked; only the diagonal block w needs the causal compare/cndmask.
  for (int jp = 0; jp < wave; ++jp)
    attn_kblock<false>(base, VT, pwave, qa, onesf, rm, rs, yacc,
                       h, jp, m0, hf, ln);
  attn_kblock<true>(base, VT, pwave, qa, onesf, rm, rs, yacc,
                    h, wave, m0, hf, ln);

  // Epilogue: normalize and write y (bf16, [B,T,384] layout)
#pragma unroll
  for (int mt = 0; mt < 2; ++mt) {
#pragma unroll
    for (int r = 0; r < 8; ++r) {
      float inv = 1.0f / rs[mt * 8 + r];
      int rowg = m0 + mt * 16 + r + hf * 8;
      size_t obase = ((size_t)bb * 256 + rowg) * 384 + h * 64;
#pragma unroll
      for (int d = 0; d < 4; ++d)
        y[obase + d * 16 + ln] = f2bf(yacc[mt * 4 + d][r] * inv);
    }
  }
}

// ---------------------------------------------------------------------------
extern "C" void kernel_launch(void* const* d_in, const int* in_sizes, int n_in,
                              void* d_out, int out_size, void* d_ws, size_t ws_size,
                              hipStream_t stream) {
  const float* x      = (const float*)d_in[0];
  const float* w_attn = (const float*)d_in[1];
  const float* b_attn = (const float*)d_in[2];
  const float* w_proj = (const float*)d_in[3];
  const float* b_proj = (const float*)d_in[4];
  float* out = (float*)d_out;

  const size_t xN   = (size_t)128 * 256 * 384;   // 12,582,912
  const size_t waN  = (size_t)1152 * 384;        //    442,368
  const size_t wpN  = (size_t)384 * 384;         //    147,456
  const size_t qkvN = (size_t)128 * 256 * 1152;  // 37,748,736

  char* p = (char*)d_ws;
  unsigned short* xb   = (unsigned short*)p; p += xN * 2;
  unsigned short* wab  = (unsigned short*)p; p += waN * 2;
  unsigned short* wpb  = (unsigned short*)p; p += wpN * 2;
  unsigned short* qkvb = (unsigned short*)p; p += qkvN * 2;
  unsigned short* yb   = (unsigned short*)p; p += xN * 2;

  cvt_f32_bf16<<<(int)((xN / 4 + 255) / 256), 256, 0, stream>>>(x, xb, (int)(xN / 4));
  cvt_f32_bf16<<<(int)((waN / 4 + 255) / 256), 256, 0, stream>>>(w_attn, wab, (int)(waN / 4));
  cvt_f32_bf16<<<(int)((wpN / 4 + 255) / 256), 256, 0, stream>>>(w_proj, wpb, (int)(wpN / 4));

  // qkv GEMM: M=32768, N=1152, K=384 -> 2048*9 wave-tiles / 8 waves = 2304 blocks
  gemm_bias_wmma<false><<<2304, 256, 0, stream>>>(xb, wab, b_attn, qkvb, nullptr,
                                                  32768, 1152, 384);
  // attention: one block per (b, head)
  attn_kernel<<<128 * 6, 256, 0, stream>>>(qkvb, yb);
  // proj GEMM: M=32768, N=384, K=384 -> 2048*3 / 8 = 768 blocks
  gemm_bias_wmma<true><<<768, 256, 0, stream>>>(yb, wpb, b_proj, nullptr, out,
                                                32768, 384, 384);
}
